// Aggregation_10814727651817
// MI455X (gfx1250) — matrix operand order repaired and checked by reading, not compiled, not run on hardware
//
#include <hip/hip_runtime.h>
#include <hip/hip_bf16.h>

// Problem dims (fixed by reference)
#define B_ 8
#define L_ 512
#define N_ 2048
#define C_ 1024
#define D_ 128

typedef __attribute__((ext_vector_type(16))) __bf16 v16bf;
typedef __attribute__((ext_vector_type(8)))  __bf16 v8bf;
typedef __attribute__((ext_vector_type(8)))  float  v8f;

__device__ __forceinline__ unsigned short f32_to_bf16_rne(float f) {
  unsigned int u = __builtin_bit_cast(unsigned int, f);
  u += 0x7FFFu + ((u >> 16) & 1u);
  return (unsigned short)(u >> 16);
}

// ---- WMMA fragment loaders (bf16, 16x16x32) -------------------------------
// A-matrix 16x32 (16-bit): lane t (row = t&15, hi = t>>4) holds
//   K = 8*hi..8*hi+7 (halves 0..7) and K = 16+8*hi..16+8*hi+7 (halves 8..15)
__device__ __forceinline__ v16bf load_a16(const unsigned short* base, int stride) {
  int t   = threadIdx.x & 31;
  int row = t & 15;
  int kb  = (t >> 4) << 3;                  // 0 or 8
  const __bf16* p = (const __bf16*)base + (size_t)row * stride + kb;
  v8bf lo = *(const v8bf*)(p);
  v8bf hi = *(const v8bf*)(p + 16);
  return __builtin_shufflevector(lo, hi, 0,1,2,3,4,5,6,7,8,9,10,11,12,13,14,15);
}

// B-matrix 32x16 (16-bit): lane t (col = t&15, hi = t>>4) holds
//   K = 8*hi..8*hi+7 (halves 0..7) and K = 16+8*hi..16+8*hi+7 (halves 8..15)
//   of column (t&15), matching the A layout.
__device__ __forceinline__ v16bf load_b16(const unsigned short* base, int stride) {
  int t   = threadIdx.x & 31;
  int col = t & 15;
  int kb  = (t >> 4) << 3;                  // 0 or 8
  const __bf16* p = (const __bf16*)base + (size_t)col * stride + kb;
  v8bf lo = *(const v8bf*)(p);
  v8bf hi = *(const v8bf*)(p + 16);
  return __builtin_shufflevector(lo, hi, 0,1,2,3,4,5,6,7,8,9,10,11,12,13,14,15);
}

// ---- Kernel 0: f32 -> bf16 conversion, 4 elems/thread ---------------------
__global__ void cvt4_kernel(const float* __restrict__ s,
                            unsigned short* __restrict__ d, int n4) {
  int i = blockIdx.x * blockDim.x + threadIdx.x;
  if (i >= n4) return;
  float4 v = ((const float4*)s)[i];
  ushort4 o;
  o.x = f32_to_bf16_rne(v.x);
  o.y = f32_to_bf16_rne(v.y);
  o.z = f32_to_bf16_rne(v.z);
  o.w = f32_to_bf16_rne(v.w);
  ((ushort4*)d)[i] = o;
}

// ---- Kernel 1: fused Q & K projection: feat[B,N,C] x W^T + b -> bf16 ------
// one wave = 16 rows x all D=128 columns of one projection (8 accum tiles).
// Explicit double buffering: next k-chunk's A + 8 B fragments are issued
// before the current chunk's WMMA burst, so >=9 loads stay in flight and the
// pre-WMMA wait cannot collapse to s_wait_loadcnt 0. waves_per_eu=1 gives the
// allocator the register budget to keep both buffers live.
__global__ __launch_bounds__(256, 1) void proj_qk_kernel(
    const unsigned short* __restrict__ feat,
    const unsigned short* __restrict__ wq, const unsigned short* __restrict__ wk,
    const float* __restrict__ bq, const float* __restrict__ bk,
    unsigned short* __restrict__ qout, unsigned short* __restrict__ kout) {
  int wave  = (int)((blockIdx.x * blockDim.x + threadIdx.x) >> 5); // 2048 waves
  int which = wave & 1;                     // 0 = Q, 1 = K
  int mt    = (wave >> 1) & 127;            // N/16 = 128
  int b     = wave >> 8;
  const unsigned short* w    = which ? wk : wq;
  const float*          bias = which ? bk : bq;
  unsigned short*       out  = which ? kout : qout;
  int m0 = mt * 16;

  const unsigned short* abase = feat + ((size_t)b * N_ + m0) * C_;
  v8f acc[8] = {};

  v16bf af = load_a16(abase, C_);
  v16bf bf[8];
#pragma unroll
  for (int dt = 0; dt < 8; ++dt)
    bf[dt] = load_b16(w + (size_t)dt * 16 * C_, C_);          // B col j = W row

  for (int c0 = 32; c0 < C_; c0 += 32) {
    // prefetch next chunk
    v16bf afn = load_a16(abase + c0, C_);
    v16bf bfn[8];
#pragma unroll
    for (int dt = 0; dt < 8; ++dt)
      bfn[dt] = load_b16(w + (size_t)dt * 16 * C_ + c0, C_);
    // compute current chunk
#pragma unroll
    for (int dt = 0; dt < 8; ++dt)
      acc[dt] = __builtin_amdgcn_wmma_f32_16x16x32_bf16(false, af, false, bf[dt],
                                                        (short)0, acc[dt],
                                                        false, false);
    af = afn;
#pragma unroll
    for (int dt = 0; dt < 8; ++dt) bf[dt] = bfn[dt];
  }
#pragma unroll
  for (int dt = 0; dt < 8; ++dt)
    acc[dt] = __builtin_amdgcn_wmma_f32_16x16x32_bf16(false, af, false, bf[dt],
                                                      (short)0, acc[dt],
                                                      false, false);

  // C/D layout: VGPR r -> M = r + 8*(t>>4), N = t&15
  int t = threadIdx.x & 31;
  int col = t & 15;
  int rbase = (t >> 4) << 3;
#pragma unroll
  for (int dt = 0; dt < 8; ++dt) {
    float bv = bias[dt * 16 + col];
    unsigned short* op = out + ((size_t)b * N_ + m0) * D_ + dt * 16 + col;
#pragma unroll
    for (int r = 0; r < 8; ++r)
      op[(size_t)(rbase + r) * D_] = f32_to_bf16_rne(acc[dt][r] + bv);
  }
}

// ---- Kernel 2: energy = Q K^T for a 16-row block, softmax in LDS ----------
// block = 256 thr (8 waves); each wave does 16 column tiles with preloaded Q
// fragments; K fragments batched before the WMMA burst; energy stripe
// [16][2048] f32 lives in dynamic LDS (128KB < 320KB/WGP)
__global__ __launch_bounds__(256, 1) void energy_softmax_kernel(
    const unsigned short* __restrict__ q, const unsigned short* __restrict__ k,
    unsigned short* __restrict__ attn) {
  extern __shared__ float sm[];
  float* E    = sm;                 // [16][N_]
  float* red  = sm + 16 * N_;       // [256]
  float* rowv = red + 256;          // [16]

  int b  = (int)(blockIdx.x >> 7);  // 128 m-tiles per batch
  int mt = (int)(blockIdx.x & 127);
  int m0 = mt * 16;
  int tid = threadIdx.x;
  int wv  = tid >> 5;

  const unsigned short* qbase = q + ((size_t)b * N_ + m0) * D_;
  v16bf qf[4];
#pragma unroll
  for (int i = 0; i < 4; ++i) qf[i] = load_a16(qbase + 32 * i, D_);

  for (int tn = wv; tn < N_ / 16; tn += 8) {      // uniform per wave
    int n0 = tn * 16;
    const unsigned short* kbase = k + ((size_t)b * N_ + n0) * D_;
    v16bf kf[4];
#pragma unroll
    for (int i = 0; i < 4; ++i)
      kf[i] = load_b16(kbase + 32 * i, D_);       // B col n = Key row n0+n
    v8f acc = {};
#pragma unroll
    for (int i = 0; i < 4; ++i)
      acc = __builtin_amdgcn_wmma_f32_16x16x32_bf16(false, qf[i], false, kf[i],
                                                    (short)0, acc, false, false);
    int t = tid & 31;
    int col = n0 + (t & 15);
    int rbase = (t >> 4) << 3;
#pragma unroll
    for (int r = 0; r < 8; ++r)
      E[(size_t)(rbase + r) * N_ + col] = acc[r];
  }
  __syncthreads();

  // softmax over n for each of the 16 rows; 16 threads per row
  int row = tid >> 4;
  int sub = tid & 15;
  float* erow = E + (size_t)row * N_;
  float mx = -3.402823466e38f;
  for (int c = sub; c < N_; c += 16) mx = fmaxf(mx, erow[c]);
  red[tid] = mx;
  __syncthreads();
  if (sub == 0) {
    float m2 = red[row * 16];
    for (int j = 1; j < 16; ++j) m2 = fmaxf(m2, red[row * 16 + j]);
    rowv[row] = m2;
  }
  __syncthreads();
  float rmax = rowv[row];
  float s = 0.f;
  for (int c = sub; c < N_; c += 16) {
    float e = __expf(erow[c] - rmax);
    erow[c] = e;
    s += e;
  }
  red[tid] = s;
  __syncthreads();
  if (sub == 0) {
    float s2 = 0.f;
    for (int j = 0; j < 16; ++j) s2 += red[row * 16 + j];
    rowv[row] = s2;
  }
  __syncthreads();
  float inv = 1.0f / rowv[row];
  unsigned short* arow = attn + ((size_t)b * N_ + (m0 + row)) * N_;
  for (int c = sub; c < N_; c += 16)
    arow[c] = f32_to_bf16_rne(erow[c] * inv);
}

// ---- Kernel 3: x_m = logits @ attention^T -> f32 out ----------------------
// out[l,m] = sum_n logits[l,n] * attn[m,n]; register-blocked 64x64 per wave:
// 16 WMMAs per 8 fragment loads, batched ahead of the WMMA burst; relaxed
// register budget keeps all 8 fragments live so the waits stagger
__global__ __launch_bounds__(256, 1) void out_gemm_kernel(
    const unsigned short* __restrict__ logits,
    const unsigned short* __restrict__ attn, float* __restrict__ out) {
  int wave = (int)((blockIdx.x * blockDim.x + threadIdx.x) >> 5); // 2048 waves
  int nt = wave & 31;                      // N/64 = 32
  int lt = (wave >> 5) & 7;                // L/64 = 8
  int b  = wave >> 8;
  int l0 = lt * 64, m0 = nt * 64;
  const unsigned short* abase = logits + ((size_t)b * L_ + l0) * N_;
  const unsigned short* bbase = attn + ((size_t)b * N_ + m0) * N_;

  v8f acc[4][4] = {};
  for (int n0 = 0; n0 < N_; n0 += 32) {
    v16bf af[4], bf[4];
#pragma unroll
    for (int i = 0; i < 4; ++i)
      af[i] = load_a16(abase + (size_t)i * 16 * N_ + n0, N_);
#pragma unroll
    for (int j = 0; j < 4; ++j)
      bf[j] = load_b16(bbase + (size_t)j * 16 * N_ + n0, N_); // col = attn row
#pragma unroll
    for (int i = 0; i < 4; ++i)
#pragma unroll
      for (int j = 0; j < 4; ++j)
        acc[i][j] = __builtin_amdgcn_wmma_f32_16x16x32_bf16(
            false, af[i], false, bf[j], (short)0, acc[i][j], false, false);
  }

  int t = threadIdx.x & 31;
  int cc = t & 15;
  int rbase = (t >> 4) << 3;
#pragma unroll
  for (int i = 0; i < 4; ++i)
#pragma unroll
    for (int j = 0; j < 4; ++j) {
      float* op = out + ((size_t)b * L_ + l0 + i * 16) * N_ + m0 + j * 16 + cc;
#pragma unroll
      for (int r = 0; r < 8; ++r)
        op[(size_t)(rbase + r) * N_] = acc[i][j][r];
    }
}

extern "C" void kernel_launch(void* const* d_in, const int* in_sizes, int n_in,
                              void* d_out, int out_size, void* d_ws, size_t ws_size,
                              hipStream_t stream) {
  const float* logits = (const float*)d_in[0];   // [B,L,N]
  const float* feat   = (const float*)d_in[1];   // [B,N,C]
  const float* Wq     = (const float*)d_in[2];   // [D,C]
  const float* bq     = (const float*)d_in[3];   // [D]
  const float* Wk     = (const float*)d_in[4];   // [D,C]
  const float* bk     = (const float*)d_in[5];   // [D]
  float* out = (float*)d_out;

  unsigned short* ws = (unsigned short*)d_ws;
  // bf16 workspace layout (element offsets; all 16B-aligned)
  size_t o_wq   = 0;
  size_t o_wk   = o_wq   + (size_t)D_ * C_;
  size_t o_feat = o_wk   + (size_t)D_ * C_;
  size_t o_log  = o_feat + (size_t)B_ * N_ * C_;
  size_t o_q    = o_log  + (size_t)B_ * L_ * N_;
  size_t o_k    = o_q    + (size_t)B_ * N_ * D_;
  size_t o_attn = o_k    + (size_t)B_ * N_ * D_;

  unsigned short* wq_b   = ws + o_wq;
  unsigned short* wk_b   = ws + o_wk;
  unsigned short* feat_b = ws + o_feat;
  unsigned short* log_b  = ws + o_log;
  unsigned short* q_b    = ws + o_q;
  unsigned short* k_b    = ws + o_k;
  unsigned short* attn_b = ws + o_attn;

  // 0) conversions (all sizes divisible by 4)
  {
    int n4; int thr = 256;
    n4 = (D_ * C_) / 4;
    cvt4_kernel<<<(n4 + thr - 1) / thr, thr, 0, stream>>>(Wq, wq_b, n4);
    cvt4_kernel<<<(n4 + thr - 1) / thr, thr, 0, stream>>>(Wk, wk_b, n4);
    n4 = (B_ * N_ * C_) / 4;
    cvt4_kernel<<<(n4 + thr - 1) / thr, thr, 0, stream>>>(feat, feat_b, n4);
    n4 = (B_ * L_ * N_) / 4;
    cvt4_kernel<<<(n4 + thr - 1) / thr, thr, 0, stream>>>(logits, log_b, n4);
  }

  // 1) Q/K projection: B*(N/16)*2 = 2048 waves -> 256 blocks
  proj_qk_kernel<<<256, 256, 0, stream>>>(feat_b, wq_b, wk_b, bq, bk, q_b, k_b);

  // 2) energy + softmax: B*(N/16) = 1024 blocks, 132160B dynamic LDS
  size_t smem = (size_t)(16 * N_ + 256 + 16) * sizeof(float);
  energy_softmax_kernel<<<1024, 256, smem, stream>>>(q_b, k_b, attn_b);

  // 3) output GEMM: B*(L/64)*(N/64) = 2048 waves -> 256 blocks
  out_gemm_kernel<<<256, 256, 0, stream>>>(log_b, attn_b, out);
}
